// KVRefiner_80556406604110
// MI455X (gfx1250) — compile-verified, matrix operand order checked by reading, not run on hardware
//
#include <hip/hip_runtime.h>
#include <hip/hip_bf16.h>

// ---------------------------------------------------------------------------
// KVRefiner for MI455X (gfx1250, wave32, WMMA + Tensor Data Mover).
// Shapes: B=2 P=128 U=64 C=1024 Dh=512 -> M=16384 rows, in_dim=3584,
// hidden=512, out_dim=2050.
// GEMMs: v_wmma_f32_16x16x32_f16 (f16 A/B, f32 accumulate).
// Weight/activation tiles staged into LDS by the TDM with double-buffered
// software pipelining (tensor_load_to_lds + s_wait_tensorcnt N).
// ---------------------------------------------------------------------------

typedef __attribute__((ext_vector_type(16))) _Float16 v16h;
typedef __attribute__((ext_vector_type(8)))  _Float16 v8h;
typedef __attribute__((ext_vector_type(8)))  float    v8f;
typedef __attribute__((ext_vector_type(4)))  unsigned int u32x4;
typedef __attribute__((ext_vector_type(4)))  int      i32x4;
typedef __attribute__((ext_vector_type(8)))  int      i32x8;

#define TAU_C      3.0f
#define BETA_MEM_C 0.2f
#define M_ROWS     16384
#define IN_DIM     3584
#define HID        512
#define OUT_DIM    2050
#define OUT_W      2112          // 33 tiles of 64, zero-padded
#define C_DIM      1024
#define DH_DIM     512

// ---------------------------------------------------------------------------
// TDM: 2D f16 tile load (global -> LDS). D# per CDNA5 ISA ch.8.
// ---------------------------------------------------------------------------
__device__ __forceinline__ void tdm_load_2d_f16(
    unsigned lds_byte_off, const _Float16* gptr,
    unsigned tensor_d0, unsigned tensor_d1,
    unsigned tile_d0, unsigned tile_d1, unsigned stride0_elems)
{
    unsigned long long ga = (unsigned long long)(uintptr_t)gptr;
    u32x4 g0;
    g0.x = 1u;                                   // count=1 (valid user D#)
    g0.y = lds_byte_off;                         // lds_addr (bytes)
    g0.z = (unsigned)ga;                         // global_addr[31:0]
    g0.w = (unsigned)((ga >> 32) & 0x01FFFFFFu) | 0x80000000u; // [56:32], type=2
    i32x8 g1;
    g1[0] = 0x00010000;                          // wg_mask=0, data_size=1 (2B)
    g1[1] = (int)((tensor_d0 & 0xFFFFu) << 16);
    g1[2] = (int)(((tensor_d0 >> 16) & 0xFFFFu) | ((tensor_d1 & 0xFFFFu) << 16));
    g1[3] = (int)(((tensor_d1 >> 16) & 0xFFFFu) | ((tile_d0 & 0xFFFFu) << 16));
    g1[4] = (int)(tile_d1 & 0xFFFFu);            // tile_dim1, tile_dim2=0
    g1[5] = (int)stride0_elems;                  // tensor_dim0_stride[31:0]
    g1[6] = 0;
    g1[7] = 0;
    i32x4 z4 = {0, 0, 0, 0};
#if __clang_major__ >= 23
    i32x8 z8 = {0, 0, 0, 0, 0, 0, 0, 0};
    __builtin_amdgcn_tensor_load_to_lds(g0, g1, z4, z4, z8, 0);
#else
    __builtin_amdgcn_tensor_load_to_lds(g0, g1, z4, z4, 0);
#endif
}

__device__ __forceinline__ unsigned lds_off_of(const void* p)
{
    // LDS flat addresses carry the LDS byte offset in addr[31:0].
    return (unsigned)(uintptr_t)p;
}

// ---------------------------------------------------------------------------
// Prep: transposed f16 weights. W1t[n][k] (512 x 3584),
// W2t[n][k] (2112 x 512, zero rows >= 2050).
// ---------------------------------------------------------------------------
__global__ __launch_bounds__(256) void kv_prep_w1t(
    const float* __restrict__ W1, _Float16* __restrict__ W1t)
{
    int idx = blockIdx.x * 256 + threadIdx.x;    // 512*3584 total
    int n = idx / IN_DIM, k = idx - n * IN_DIM;
    W1t[idx] = (_Float16)W1[(size_t)k * HID + n];
}

__global__ __launch_bounds__(256) void kv_prep_w2t(
    const float* __restrict__ W2, _Float16* __restrict__ W2t)
{
    int idx = blockIdx.x * 256 + threadIdx.x;    // 2112*512 total
    int n = idx / HID, k = idx - n * HID;
    W2t[idx] = (n < OUT_DIM) ? (_Float16)W2[(size_t)k * OUT_DIM + n]
                             : (_Float16)0.0f;
}

// ---------------------------------------------------------------------------
// Kernel 1: per-row LayerNorm statistics.
// ---------------------------------------------------------------------------
__global__ __launch_bounds__(256) void kv_stats(
    const float* __restrict__ q, const float* __restrict__ K,
    const float* __restrict__ V, const float* __restrict__ r,
    float2* __restrict__ stats)
{
    const int i = blockIdx.x;
    const int tid = threadIdx.x;
    const float* qrow = q + (size_t)(i >> 6)  * C_DIM;
    const float* Krow = K + (size_t)i * C_DIM;
    const float* Vrow = V + (size_t)i * C_DIM;
    const float* rrow = r + (size_t)(i >> 13) * DH_DIM;

    float s = 0.f, sq = 0.f;
    for (int j = tid; j < C_DIM; j += 256) {
        float a = qrow[j], b = Krow[j], c = Vrow[j];
        s += a + b + c; sq += a * a + b * b + c * c;
    }
    for (int j = tid; j < DH_DIM; j += 256) {
        float a = rrow[j]; s += a; sq += a * a;
    }
    __shared__ float rs[256], rq[256];
    rs[tid] = s; rq[tid] = sq; __syncthreads();
    for (int st = 128; st > 0; st >>= 1) {
        if (tid < st) { rs[tid] += rs[tid + st]; rq[tid] += rq[tid + st]; }
        __syncthreads();
    }
    if (tid == 0) {
        float mu  = rs[0] * (1.0f / IN_DIM);
        float var = rq[0] * (1.0f / IN_DIM) - mu * mu;
        stats[i] = make_float2(mu, rsqrtf(fmaxf(var, 0.f) + 1e-5f));
    }
}

__device__ __forceinline__ float gelu_exact(float x)
{
    return 0.5f * x * (1.0f + erff(x * 0.70710678118654752f));
}

// ---------------------------------------------------------------------------
// Kernel 2: h = gelu(LN(inp) @ W1 + b1) -> h16 [16384 x 512] f16.
// A staged by VALU (normalize+cvt), B staged by TDM, double-buffered.
// ---------------------------------------------------------------------------
__global__ __launch_bounds__(256) void kv_gemm1(
    const float* __restrict__ q, const float* __restrict__ K,
    const float* __restrict__ V, const float* __restrict__ r,
    const float* __restrict__ gamma, const float* __restrict__ beta,
    const _Float16* __restrict__ W1t, const float* __restrict__ b1,
    const float2* __restrict__ stats, _Float16* __restrict__ h16)
{
    __shared__ __align__(16) _Float16 As[2][128][32];
    __shared__ __align__(16) _Float16 Bs[2][64][32];

    const int tid  = threadIdx.x;
    const int lane = tid & 31;
    const int wave = tid >> 5;
    const int wm   = wave >> 1;
    const int wn   = wave & 1;
    const int mBase = blockIdx.y * 128;
    const int nBase = blockIdx.x * 64;

    const int ar = tid >> 1;
    const int ak = (tid & 1) * 16;
    const int gi = mBase + ar;
    const float2 st = stats[gi];
    const float mu = st.x, rstd = st.y;
    const float* qrow = q + (size_t)(gi >> 6)  * C_DIM;
    const float* Krow = K + (size_t)gi * C_DIM;
    const float* Vrow = V + (size_t)gi * C_DIM;
    const float* rrow = r + (size_t)(gi >> 13) * DH_DIM;

    const int kb  = (lane & 16) ? 8 : 0;
    const int l15 = lane & 15;
    const unsigned bsOff0 = lds_off_of(&Bs[0][0][0]);
    const unsigned bsOff1 = lds_off_of(&Bs[1][0][0]);
    const _Float16* w1base = W1t + (size_t)nBase * IN_DIM;

    v8f acc[2][2] = {};

    // prologue: kick B tile for k0 = 0 into buffer 0
    if (tid < 32)
        tdm_load_2d_f16(bsOff0, w1base, IN_DIM, HID, 32, 64, IN_DIM);

    int phase = 0;
    for (int k0 = 0; k0 < IN_DIM; k0 += 32) {
        // ---- stage A tile into As[phase] (uniform region select) ----
        const float* srcRow;
        if      (k0 < 1024) srcRow = qrow + k0;
        else if (k0 < 2048) srcRow = Krow + (k0 - 1024);
        else if (k0 < 3072) srcRow = Vrow + (k0 - 2048);
        else                srcRow = rrow + (k0 - 3072);
        v8h frag_lo, frag_hi;
        #pragma unroll
        for (int jv = 0; jv < 4; ++jv) {
            float4 x  = *(const float4*)(srcRow + ak + jv * 4);
            float4 gm = *(const float4*)(gamma + k0 + ak + jv * 4);
            float4 bt = *(const float4*)(beta  + k0 + ak + jv * 4);
            float n0 = (x.x - mu) * rstd * gm.x + bt.x;
            float n1 = (x.y - mu) * rstd * gm.y + bt.y;
            float n2 = (x.z - mu) * rstd * gm.z + bt.z;
            float n3 = (x.w - mu) * rstd * gm.w + bt.w;
            if (jv < 2) {
                frag_lo[jv * 4 + 0] = (_Float16)n0; frag_lo[jv * 4 + 1] = (_Float16)n1;
                frag_lo[jv * 4 + 2] = (_Float16)n2; frag_lo[jv * 4 + 3] = (_Float16)n3;
            } else {
                frag_hi[(jv - 2) * 4 + 0] = (_Float16)n0; frag_hi[(jv - 2) * 4 + 1] = (_Float16)n1;
                frag_hi[(jv - 2) * 4 + 2] = (_Float16)n2; frag_hi[(jv - 2) * 4 + 3] = (_Float16)n3;
            }
        }
        *(v8h*)&As[phase][ar][ak]     = frag_lo;
        *(v8h*)&As[phase][ar][ak + 8] = frag_hi;

        // ---- prefetch next B tile into the other buffer, wait for current ----
        if (tid < 32) {
            if (k0 + 32 < IN_DIM) {
                tdm_load_2d_f16(phase ? bsOff0 : bsOff1,
                                w1base + (k0 + 32),
                                IN_DIM, HID, 32, 64, IN_DIM);
                __builtin_amdgcn_s_wait_tensorcnt(1);   // current tile done
            } else {
                __builtin_amdgcn_s_wait_tensorcnt(0);
            }
        }
        __syncthreads();

        // ---- fragments + WMMA on buffer[phase] ----
        v16h afrag[2], bfrag[2];
        #pragma unroll
        for (int t = 0; t < 2; ++t) {
            const _Float16* ap = &As[phase][wm * 32 + t * 16 + l15][kb];
            const _Float16* bp = &Bs[phase][wn * 32 + t * 16 + l15][kb];
            #pragma unroll
            for (int e = 0; e < 8; ++e) {
                afrag[t][e] = ap[e];  afrag[t][e + 8] = ap[e + 16];
                bfrag[t][e] = bp[e];  bfrag[t][e + 8] = bp[e + 16];
            }
        }
        #pragma unroll
        for (int tm = 0; tm < 2; ++tm)
            #pragma unroll
            for (int tn = 0; tn < 2; ++tn)
                acc[tm][tn] = __builtin_amdgcn_wmma_f32_16x16x32_f16(
                    false, afrag[tm], false, bfrag[tn],
                    (short)0, acc[tm][tn], false, false);
        __syncthreads();
        phase ^= 1;
    }

    const int mOff = (lane & 16) ? 8 : 0;
    #pragma unroll
    for (int tm = 0; tm < 2; ++tm)
        #pragma unroll
        for (int tn = 0; tn < 2; ++tn)
            #pragma unroll
            for (int v = 0; v < 8; ++v) {
                int row = mBase + wm * 32 + tm * 16 + mOff + v;
                int col = nBase + wn * 32 + tn * 16 + l15;
                float hv = gelu_exact(acc[tm][tn][v] + b1[col]);
                h16[(size_t)row * HID + col] = (_Float16)hv;
            }
}

// ---------------------------------------------------------------------------
// Kernel 3: out = h @ W2 + b2 [16384 x 2050], padded to OUT_W.
// A and B tiles both staged by TDM, double-buffered (2 DMAs in flight).
// ---------------------------------------------------------------------------
__global__ __launch_bounds__(256) void kv_gemm2(
    const _Float16* __restrict__ h16, const _Float16* __restrict__ W2t,
    const float* __restrict__ b2, float* __restrict__ outb)
{
    __shared__ __align__(16) _Float16 As[2][128][32];
    __shared__ __align__(16) _Float16 Bs[2][64][32];

    const int tid  = threadIdx.x;
    const int lane = tid & 31;
    const int wave = tid >> 5;
    const int wm   = wave >> 1;
    const int wn   = wave & 1;
    const int mBase = blockIdx.y * 128;
    const int nBase = blockIdx.x * 64;

    const int kb  = (lane & 16) ? 8 : 0;
    const int l15 = lane & 15;
    const unsigned asOff[2] = { lds_off_of(&As[0][0][0]), lds_off_of(&As[1][0][0]) };
    const unsigned bsOff[2] = { lds_off_of(&Bs[0][0][0]), lds_off_of(&Bs[1][0][0]) };
    const _Float16* habase = h16 + (size_t)mBase * HID;
    const _Float16* w2base = W2t + (size_t)nBase * HID;

    v8f acc[2][2] = {};

    if (tid < 32) {
        tdm_load_2d_f16(asOff[0], habase, HID, M_ROWS, 32, 128, HID);
        tdm_load_2d_f16(bsOff[0], w2base, HID, OUT_W, 32, 64, HID);
    }

    int phase = 0;
    for (int k0 = 0; k0 < HID; k0 += 32) {
        if (tid < 32) {
            if (k0 + 32 < HID) {
                tdm_load_2d_f16(asOff[phase ^ 1], habase + (k0 + 32),
                                HID, M_ROWS, 32, 128, HID);
                tdm_load_2d_f16(bsOff[phase ^ 1], w2base + (k0 + 32),
                                HID, OUT_W, 32, 64, HID);
                __builtin_amdgcn_s_wait_tensorcnt(2);   // current pair done
            } else {
                __builtin_amdgcn_s_wait_tensorcnt(0);
            }
        }
        __syncthreads();

        v16h afrag[2], bfrag[2];
        #pragma unroll
        for (int t = 0; t < 2; ++t) {
            const _Float16* ap = &As[phase][wm * 32 + t * 16 + l15][kb];
            const _Float16* bp = &Bs[phase][wn * 32 + t * 16 + l15][kb];
            #pragma unroll
            for (int e = 0; e < 8; ++e) {
                afrag[t][e] = ap[e];  afrag[t][e + 8] = ap[e + 16];
                bfrag[t][e] = bp[e];  bfrag[t][e + 8] = bp[e + 16];
            }
        }
        #pragma unroll
        for (int tm = 0; tm < 2; ++tm)
            #pragma unroll
            for (int tn = 0; tn < 2; ++tn)
                acc[tm][tn] = __builtin_amdgcn_wmma_f32_16x16x32_f16(
                    false, afrag[tm], false, bfrag[tn],
                    (short)0, acc[tm][tn], false, false);
        __syncthreads();
        phase ^= 1;
    }

    const int mOff = (lane & 16) ? 8 : 0;
    #pragma unroll
    for (int tm = 0; tm < 2; ++tm)
        #pragma unroll
        for (int tn = 0; tn < 2; ++tn)
            #pragma unroll
            for (int v = 0; v < 8; ++v) {
                int row = mBase + wm * 32 + tm * 16 + mOff + v;
                int col = nBase + wn * 32 + tn * 16 + l15;
                float o = acc[tm][tn][v] + (col < OUT_DIM ? b2[col] : 0.0f);
                outb[(size_t)row * OUT_W + col] = o;
            }
}

// ---------------------------------------------------------------------------
// Kernel 4: per-row epilogue.
// ---------------------------------------------------------------------------
__device__ __forceinline__ float blk_sum(float v, volatile float* red, int tid)
{
    red[tid] = v; __syncthreads();
    for (int st = 128; st > 0; st >>= 1) {
        if (tid < st) red[tid] += red[tid + st];
        __syncthreads();
    }
    float r = red[0]; __syncthreads();
    return r;
}

__global__ __launch_bounds__(256) void kv_epilogue(
    const float* __restrict__ outb, const float* __restrict__ K,
    const float* __restrict__ V, const float* __restrict__ r,
    float* __restrict__ Kout, float* __restrict__ Vout,
    float* __restrict__ partial)
{
    const int i = blockIdx.x;
    const int tid = threadIdx.x;
    __shared__ float red[256];

    const float* orow = outb + (size_t)i * OUT_W;
    const float s = orow[2048];
    const float g = orow[2049];
    const float* Krow = K + (size_t)i * C_DIM;
    const float* Vrow = V + (size_t)i * C_DIM;
    const float* rrow = r + (size_t)(i >> 13) * DH_DIM;

    float kv[4], vv[4];
    float dsum = 0.f, ksum = 0.f, vsum = 0.f;
    #pragma unroll
    for (int t = 0; t < 4; ++t) {
        int c = tid + t * 256;
        float dK = orow[c];
        float dV = orow[C_DIM + c];
        dsum += dK * dK + dV * dV;
        float kn = Krow[c] + g * dK;
        kv[t] = kn; ksum += kn * kn;
        float vn = Vrow[c] * (1.f + s) + g * dV;
        vv[t] = vn; vsum += vn * vn;
    }

    float kn2 = blk_sum(ksum, red, tid);
    float kscale = fminf(TAU_C / fmaxf(sqrtf(kn2), 1e-6f), 1.0f);
    #pragma unroll
    for (int t = 0; t < 4; ++t)
        Kout[(size_t)i * C_DIM + tid + t * 256] = kv[t] * kscale;

    float vn2 = blk_sum(vsum, red, tid);
    float vscale = fminf(TAU_C / fmaxf(sqrtf(vn2), 1e-6f), 1.0f);
    float v2sum = 0.f;
    #pragma unroll
    for (int t = 0; t < 4; ++t) {
        int c = tid + t * 256;
        float x = vv[t] * vscale + BETA_MEM_C * g * rrow[c & (DH_DIM - 1)];
        vv[t] = x; v2sum += x * x;
    }
    float v2n2 = blk_sum(v2sum, red, tid);
    float v2scale = fminf(TAU_C / fmaxf(sqrtf(v2n2), 1e-6f), 1.0f);
    #pragma unroll
    for (int t = 0; t < 4; ++t)
        Vout[(size_t)i * C_DIM + tid + t * 256] = vv[t] * v2scale;

    float dtot = blk_sum(dsum, red, tid);
    if (tid == 0) partial[i] = dtot;
}

// ---------------------------------------------------------------------------
// Kernel 5: deterministic final reduction of delta_reg.
// ---------------------------------------------------------------------------
__global__ __launch_bounds__(256) void kv_delta(
    const float* __restrict__ partial, float* __restrict__ delta)
{
    __shared__ float red[256];
    const int tid = threadIdx.x;
    float s = 0.f;
    for (int j = 0; j < 64; ++j) s += partial[tid + j * 256];  // fixed order
    red[tid] = s; __syncthreads();
    for (int st = 128; st > 0; st >>= 1) {
        if (tid < st) red[tid] += red[tid + st];
        __syncthreads();
    }
    if (tid == 0) *delta = red[0] * (1.0f / 16777216.0f);      // /(16384*1024)
}

// ---------------------------------------------------------------------------
extern "C" void kernel_launch(void* const* d_in, const int* in_sizes, int n_in,
                              void* d_out, int out_size, void* d_ws, size_t ws_size,
                              hipStream_t stream)
{
    const float* q     = (const float*)d_in[0];
    const float* K     = (const float*)d_in[1];
    const float* V     = (const float*)d_in[2];
    const float* r     = (const float*)d_in[3];
    const float* gamma = (const float*)d_in[4];
    const float* beta  = (const float*)d_in[5];
    const float* W1    = (const float*)d_in[6];
    const float* b1    = (const float*)d_in[7];
    const float* W2    = (const float*)d_in[8];
    const float* b2    = (const float*)d_in[9];

    // workspace layout (bytes)
    char* ws = (char*)d_ws;
    size_t off = 0;
    float2*   stats   = (float2*)(ws + off);   off += (size_t)M_ROWS * 8;
    _Float16* h16     = (_Float16*)(ws + off); off += (size_t)M_ROWS * HID * 2;
    float*    outb    = (float*)(ws + off);    off += (size_t)M_ROWS * OUT_W * 4;
    float*    partial = (float*)(ws + off);    off += (size_t)M_ROWS * 4;
    _Float16* W1t     = (_Float16*)(ws + off); off += (size_t)HID * IN_DIM * 2;
    _Float16* W2t     = (_Float16*)(ws + off); off += (size_t)OUT_W * HID * 2;

    float* Kout  = (float*)d_out;
    float* Vout  = Kout + (size_t)M_ROWS * C_DIM;
    float* delta = Vout + (size_t)M_ROWS * C_DIM;

    kv_prep_w1t<<<dim3((HID * IN_DIM) / 256), dim3(256), 0, stream>>>(W1, W1t);
    kv_prep_w2t<<<dim3((OUT_W * HID) / 256), dim3(256), 0, stream>>>(W2, W2t);
    kv_stats   <<<dim3(M_ROWS), dim3(256), 0, stream>>>(q, K, V, r, stats);
    kv_gemm1   <<<dim3(HID / 64, M_ROWS / 128), dim3(256), 0, stream>>>(
                   q, K, V, r, gamma, beta, W1t, b1, stats, h16);
    kv_gemm2   <<<dim3(OUT_W / 64, M_ROWS / 128), dim3(256), 0, stream>>>(
                   h16, W2t, b2, outb);
    kv_epilogue<<<dim3(M_ROWS), dim3(256), 0, stream>>>(
                   outb, K, V, r, Kout, Vout, partial);
    kv_delta   <<<dim3(1), dim3(256), 0, stream>>>(partial, delta);
}
